// TorusConv3D_11115375362848
// MI455X (gfx1250) — matrix operand order repaired and checked by reading, not compiled, not run on hardware
//
#include <hip/hip_runtime.h>
#include <hip/hip_bf16.h>

typedef __attribute__((ext_vector_type(16))) _Float16 v16h;
typedef __attribute__((ext_vector_type(8)))  _Float16 v8h;
typedef __attribute__((ext_vector_type(8)))  float    v8f;
typedef __attribute__((ext_vector_type(4)))  unsigned int v4u;
typedef __attribute__((ext_vector_type(8)))  int      v8i;
typedef __attribute__((ext_vector_type(4)))  int      v4i;

// Geometry: B=64, X=Y=Z=8, C=64, F=64
// in16 layout : [b][x][y][z][c]   strides (halves): c=1, z=64, y=512, x=4096, b=32768
// k16  layout : [t][f][c]         strides (halves): c=1, f=64, t=4096  (t = tx*64+ty*8+tz)
// out  layout : [b][x][y][z][f]   strides (floats): f=1, z=64, y=512, x=4096, b=32768

#define IN_ROW   520                         // 512 halves + 4-dword TDM pad (260 dw stride)
#define IN_SLABH (64 * IN_ROW)               // 33280 halves
#define K_ROW    72                          // 64 halves + 4-dword TDM pad (36 dw stride)
#define K_SLABH  (8 * 64 * K_ROW)            // 36864 halves
#define BUFH     (IN_SLABH + K_SLABH)        // 70144 halves per pipeline buffer
#define LDS_BYTES (2 * BUFH * 2)             // 280576 bytes (double buffered, <= 320K)

#if defined(__has_builtin)
#if __has_builtin(__builtin_amdgcn_tensor_load_to_lds)
#define HAVE_TDM 1
#endif
#endif

__global__ void cvt_f32_to_f16(const float* __restrict__ src,
                               _Float16* __restrict__ dst, int n) {
    int i = blockIdx.x * blockDim.x + threadIdx.x;
    if (i < n) dst[i] = (_Float16)src[i];
}

__device__ inline v16h make16(v8h lo, v8h hi) {
    return __builtin_shufflevector(lo, hi, 0, 1, 2, 3, 4, 5, 6, 7,
                                   8, 9, 10, 11, 12, 13, 14, 15);
}

// A fragment: 16x32 f16, rows = batch (LDS [b][IN_ROW], slice at zz*64).
__device__ inline v16h load_a(const _Float16* lin, int zz, int btile, int lane, int cs) {
    const _Float16* row = lin + (btile * 16 + (lane & 15)) * IN_ROW + zz * 64;
    int h = (lane >> 4) * 8;
    v8h lo = *(const v8h*)(row + cs + h);
    v8h hi = *(const v8h*)(row + cs + 16 + h);
    return make16(lo, hi);
}

// B fragment: 32x16 f16, cols = filters (LDS [tz][f][K_ROW], native kernel rows).
__device__ inline v16h load_b(const _Float16* kb, int ftile, int lane, int cs) {
    const _Float16* row = kb + (ftile * 16 + (lane & 15)) * K_ROW;
    int c0 = cs + (lane >> 4) * 8;
    v8h lo = *(const v8h*)(row + c0);
    v8h hi = *(const v8h*)(row + c0 + 16);
    return make16(lo, hi);
}

#ifdef HAVE_TDM
// Issue one 2D TDM tile load (global -> LDS) per CDNA5 D# layout.
// Units: data_size = 2 bytes (f16). pad fields: interval code (2<<code dwords),
// amount code (code+1 dwords).
__device__ __forceinline__ void tdm_load_2d(unsigned int lds_off, const void* gptr,
                                            unsigned int tdim0, unsigned int tdim1,
                                            unsigned int tile0, unsigned int tile1,
                                            unsigned int stride0,
                                            unsigned int pad_icode, unsigned int pad_acode) {
    unsigned long long ga = (unsigned long long)(uintptr_t)gptr;
    v4u g0;
    g0.x = 1u;                                                    // count=1, user desc
    g0.y = lds_off;                                               // LDS byte address
    g0.z = (unsigned int)ga;                                      // global_addr[31:0]
    g0.w = (unsigned int)((ga >> 32) & 0x01FFFFFFu) | (2u << 30); // addr[56:32] | type=2
    v8i g1;
    g1[0] = (int)((1u << 16) |                 // data_size = 2 bytes
                  (1u << 20) |                 // pad_enable
                  (pad_icode << 22) | (pad_acode << 25));
    g1[1] = (int)((tdim0 & 0xFFFFu) << 16);                         // tensor_dim0[15:0]
    g1[2] = (int)((tdim0 >> 16) | ((tdim1 & 0xFFFFu) << 16));       // tdim0 hi | tdim1 lo
    g1[3] = (int)((tdim1 >> 16) | (tile0 << 16));                   // tdim1 hi | tile_dim0
    g1[4] = (int)tile1;                                             // tile_dim1 (tile_dim2=0)
    g1[5] = (int)stride0;                                           // tensor_dim0_stride lo
    g1[6] = 0;                                                      // stride0 hi | stride1 lo
    g1[7] = 0;
    v4i z4 = {0, 0, 0, 0};
    v8i z8 = {0, 0, 0, 0, 0, 0, 0, 0};
    __builtin_amdgcn_tensor_load_to_lds(g0, g1, z4, z4, z8, 0);
}
#endif

__global__ __launch_bounds__(512, 1)
void torus_conv_wmma(const _Float16* __restrict__ In16,
                     const _Float16* __restrict__ K16,
                     const float* __restrict__ bias,
                     float* __restrict__ out) {
    extern __shared__ _Float16 lds[];

    const int tid   = threadIdx.x;
    const int lane  = tid & 31;
    const int wave  = tid >> 5;          // 0..15
    const int btile = wave & 3;
    const int ftile = wave >> 2;

    const int wg = blockIdx.x;           // 256 workgroups
    const int z0 = (wg & 3) * 2;         // z pair
    const int y  = (wg >> 2) & 7;
    const int x  = wg >> 5;

#ifdef HAVE_TDM
    const unsigned int lds_base = (unsigned int)(uintptr_t)(void*)lds;
    // Issue both TDM slab loads for shift index i into pipeline buffer buf.
    auto issue = [&](int i, int buf) {
        const int tx = i >> 3;
        const int ty = i & 7;
        const int xx = (x - tx) & 7;
        const int yy = (y - ty) & 7;
        const unsigned int base = lds_base + (unsigned int)buf * (BUFH * 2);
        // input slab: 64 rows (b) x 512 halves ([z][c]), row stride 32768 halves,
        // LDS pad: 4 dwords after each 256 dwords -> IN_ROW=520 halves
        tdm_load_2d(base, In16 + (size_t)xx * 4096 + (size_t)yy * 512,
                    512, 64, 512, 64, 32768, /*pad 256dw*/ 7, /*+4dw*/ 3);
        // kernel slab: 512 rows (tz*f) x 64 halves, contiguous, pad 4 dwords
        // after each 32 dwords -> K_ROW=72 halves
        tdm_load_2d(base + IN_SLABH * 2, K16 + (size_t)i * 32768,
                    64, 512, 64, 512, 64, /*pad 32dw*/ 4, /*+4dw*/ 3);
    };
#endif

    v8f acc0 = {};                       // z = z0
    v8f acc1 = {};                       // z = z0+1

#ifdef HAVE_TDM
    if (wave == 0) issue(0, 0);
#endif

    for (int txy = 0; txy < 64; ++txy) {
        const int buf = txy & 1;
        const _Float16* lin = lds + buf * BUFH;      // [b 64][IN_ROW]
        const _Float16* lk  = lin + IN_SLABH;        // [tz 8][f 64][K_ROW]

#ifdef HAVE_TDM
        if (wave == 0) __builtin_amdgcn_s_wait_tensorcnt(0);  // slab txy landed
        __syncthreads();   // publish slab txy; all waves done with other buffer
        if (wave == 0 && txy + 1 < 64) issue(txy + 1, buf ^ 1); // overlap with compute
#else
        {   // manual staging fallback (no TDM builtin)
            const int tx = txy >> 3;
            const int ty = txy & 7;
            const int xx = (x - tx) & 7;
            const int yy = (y - ty) & 7;
            __syncthreads();
            const int s_b  = tid >> 3;   // 0..63
            const int s_zz = tid & 7;    // 0..7
            const uint4* src = (const uint4*)(In16 +
                ((size_t)s_b * 32768 + (size_t)xx * 4096 + (size_t)yy * 512 + s_zz * 64));
            uint4* dst = (uint4*)((_Float16*)lin + (size_t)s_b * IN_ROW + s_zz * 64);
#pragma unroll
            for (int j = 0; j < 4; ++j) dst[j] = src[j];
            const int s_tz = tid >> 6;   // 0..7
            const int s_f  = tid & 63;   // 0..63
            const uint4* ks = (const uint4*)(K16 +
                ((size_t)(txy * 8 + s_tz) * 4096 + s_f * 64));
            uint4* kd = (uint4*)((_Float16*)lk + (s_tz * 64 + s_f) * K_ROW);
#pragma unroll
            for (int j = 0; j < 8; ++j) kd[j] = ks[j];
            __syncthreads();
        }
#endif

#pragma unroll
        for (int tz = 0; tz < 8; ++tz) {
            const _Float16* kb = lk + tz * 64 * K_ROW;
            v16h b0 = load_b(kb, ftile, lane, 0);
            v16h b1 = load_b(kb, ftile, lane, 32);

            const int zz0 = (z0 - tz) & 7;
            const int zz1 = (z0 + 1 - tz) & 7;

            v16h a00 = load_a(lin, zz0, btile, lane, 0);
            v16h a01 = load_a(lin, zz0, btile, lane, 32);
            v16h a10 = load_a(lin, zz1, btile, lane, 0);
            v16h a11 = load_a(lin, zz1, btile, lane, 32);

            acc0 = __builtin_amdgcn_wmma_f32_16x16x32_f16(false, a00, false, b0,
                                                          (short)0, acc0, false, false);
            acc0 = __builtin_amdgcn_wmma_f32_16x16x32_f16(false, a01, false, b1,
                                                          (short)0, acc0, false, false);
            acc1 = __builtin_amdgcn_wmma_f32_16x16x32_f16(false, a10, false, b0,
                                                          (short)0, acc1, false, false);
            acc1 = __builtin_amdgcn_wmma_f32_16x16x32_f16(false, a11, false, b1,
                                                          (short)0, acc1, false, false);
        }
    }

    // epilogue: C/D layout -> lanes 0-15: M=r, N=lane ; lanes 16-31: M=r+8, N=lane-16
    const int fl = lane & 15;
    const int hi = lane >> 4;
    const float bv = bias[ftile * 16 + fl];
    const size_t obase = (size_t)x * 4096 + (size_t)y * 512 + ftile * 16 + fl;
#pragma unroll
    for (int r = 0; r < 8; ++r) {
        const int b = btile * 16 + r + hi * 8;
        const size_t o = (size_t)b * 32768 + obase;
        out[o + (size_t)z0 * 64]       = acc0[r] + bv;
        out[o + (size_t)(z0 + 1) * 64] = acc1[r] + bv;
    }
}

extern "C" void kernel_launch(void* const* d_in, const int* in_sizes, int n_in,
                              void* d_out, int out_size, void* d_ws, size_t ws_size,
                              hipStream_t stream) {
    const float* in_f32  = (const float*)d_in[0];  // 64*8*8*8*64
    const float* ker_f32 = (const float*)d_in[1];  // 8*8*8*64*64
    const float* bias    = (const float*)d_in[2];  // 64
    float* out = (float*)d_out;

    const int N_IN  = 64 * 8 * 8 * 8 * 64;  // 2097152
    const int N_KER = 8 * 8 * 8 * 64 * 64;  // 2097152

    _Float16* In16 = (_Float16*)d_ws;
    _Float16* K16  = (_Float16*)((char*)d_ws + (size_t)N_IN * sizeof(_Float16));

    // Opt in to >64KB dynamic LDS (non-stream API; safe under graph capture).
    static_cast<void>(hipFuncSetAttribute(
        reinterpret_cast<const void*>(&torus_conv_wmma),
        hipFuncAttributeMaxDynamicSharedMemorySize, LDS_BYTES));

    cvt_f32_to_f16<<<(N_IN + 255) / 256, 256, 0, stream>>>(in_f32, In16, N_IN);
    cvt_f32_to_f16<<<(N_KER + 255) / 256, 256, 0, stream>>>(ker_f32, K16, N_KER);

    torus_conv_wmma<<<256, 512, LDS_BYTES, stream>>>(In16, K16, bias, out);
}
